// ResidualGNNLayer_32676111188087
// MI455X (gfx1250) — compile-verified, matrix operand order checked by reading, not run on hardware
//
#include <hip/hip_runtime.h>

// ---------------------------------------------------------------------------
// ResidualGNNLayer for MI455X (gfx1250, wave32, WMMA).
// Pipeline: zero ws -> pack W1/W2 into bf16 WMMA B-fragments -> edge scatter
// (f32 global atomics, L2-resident) -> fused feat-build + GEMM1(relu) +
// GEMM2 + residual + LayerNorm.
// Each 256-thread block now owns a 32-row node tile (2 M-tiles) so every
// global B-fragment load is reused by 2 WMMAs (M register blocking), halving
// weight refetch traffic and giving each load an extra WMMA to hide behind.
// ---------------------------------------------------------------------------

typedef __bf16 bf16_t;
typedef __bf16 v8bf  __attribute__((ext_vector_type(8)));
typedef __bf16 v16bf __attribute__((ext_vector_type(16)));
typedef float  v8f   __attribute__((ext_vector_type(8)));

#define DD 128              // node feature dim
#define HH 512              // hidden dim
#define KF 288              // feat K (258) padded to multiple of 32
#define ROWS 32             // node rows per block (2 M-tiles)
#define NT1 (HH / 16)       // 32 N-tiles in GEMM1
#define KS1 (KF / 32)       // 9  K-steps in GEMM1
#define NT2 (DD / 16)       // 8  N-tiles in GEMM2
#define KS2 (HH / 32)       // 16 K-steps in GEMM2
#define W1P_ELEMS (NT1 * KS1 * 32 * 16)   // 147456 bf16
#define W2P_ELEMS (NT2 * KS2 * 32 * 16)   // 65536  bf16

// ---------------------------------------------------------------------------
__global__ __launch_bounds__(256) void zero_f32(float* __restrict__ p, int n) {
    int i = blockIdx.x * blockDim.x + threadIdx.x;
    if (i < n) p[i] = 0.0f;
}

// Pack W1 [258x512] and W2 [512x128] (f32 row-major) into bf16 per-lane
// B-fragment layout: entry ((ntile*KS + ks)*32 + lane)*16 + j, where the 16
// bf16 values per lane are K = {half*8+0..7, 16+half*8+0..7} for column
// n = ntile*16 + (lane&15), half = lane>>4 (matches 16-bit WMMA operand map).
__global__ __launch_bounds__(256) void pack_weights(
    const float* __restrict__ W1, const float* __restrict__ W2,
    bf16_t* __restrict__ w1p, bf16_t* __restrict__ w2p) {
    int idx = blockIdx.x * blockDim.x + threadIdx.x;
    if (idx < W1P_ELEMS) {
        int j    = idx & 15;
        int lane = (idx >> 4) & 31;
        int ks   = (idx >> 9) % KS1;     // idx>>9 == ntile*KS1 + ks
        int half = lane >> 4;
        int nt   = idx / (16 * 32 * KS1);
        int jj   = (j < 8) ? (half * 8 + j) : (16 + half * 8 + (j - 8));
        int k    = ks * 32 + jj;
        int n    = nt * 16 + (lane & 15);
        float v  = (k < 2 * DD + 2) ? W1[k * HH + n] : 0.0f;  // pad K 258->288
        w1p[idx] = (bf16_t)v;
    } else if (idx < W1P_ELEMS + W2P_ELEMS) {
        int e    = idx - W1P_ELEMS;
        int j    = e & 15;
        int lane = (e >> 4) & 31;
        int ks   = (e >> 9) % KS2;
        int half = lane >> 4;
        int nt   = e / (16 * 32 * KS2);
        int jj   = (j < 8) ? (half * 8 + j) : (16 + half * 8 + (j - 8));
        int k    = ks * 32 + jj;
        int n    = nt * 16 + (lane & 15);
        w2p[e]   = (bf16_t)W2[k * DD + n];
    }
}

// Edge scatter: 32 lanes per edge, float4 per lane -> 4 f32 atomics.
// x (25.6MB) and agg (25.6MB) both fit in the 192MB L2, so these atomics and
// gathers are L2-resident rather than HBM-bound.
__global__ __launch_bounds__(256) void scatter_edges(
    const float* __restrict__ x, const long long* __restrict__ ei,
    const float* __restrict__ ed, float* __restrict__ agg,
    float* __restrict__ dsum, int E_) {
    long long t = (long long)blockIdx.x * blockDim.x + threadIdx.x;
    long long e = t >> 5;
    int lane    = (int)(t & 31);
    if (e >= E_) return;
    long long s = ei[e];                  // src node (scatter target)
    long long d = ei[(long long)E_ + e];  // dst node (gather source)
    const float4 v = *(const float4*)(x + d * DD + lane * 4);
    float* a = agg + s * DD + lane * 4;
    atomicAdd(a + 0, v.x);
    atomicAdd(a + 1, v.y);
    atomicAdd(a + 2, v.z);
    atomicAdd(a + 3, v.w);
    if (lane == 0) atomicAdd(dsum + s, ed[e]);
}

// ---------------------------------------------------------------------------
// Fused: build feat tile -> GEMM1 (bf16 WMMA, relu) -> GEMM2 -> residual -> LN.
// One block = 32 node rows (2 M-tiles), 8 waves.
// LDS: 18432 + 32768 + 16384 = 67.6 KB  (~4 blocks per 320KB WGP).
__global__ __launch_bounds__(256) void fused_gnn(
    const float* __restrict__ x, const float* __restrict__ b1,
    const float* __restrict__ b2, const float* __restrict__ gamma,
    const float* __restrict__ beta, const float* __restrict__ degrees,
    const float* __restrict__ agg, const float* __restrict__ dsum,
    const bf16_t* __restrict__ w1p, const bf16_t* __restrict__ w2p,
    float* __restrict__ out, int Ntot) {
    __shared__ __align__(16) bf16_t featbf[ROWS * KF];  // feat tile, bf16
    __shared__ __align__(16) bf16_t hlds[ROWS * HH];    // hidden tile, bf16
    __shared__ __align__(16) float  ylds[ROWS * DD];    // pre-LN tile, f32

    const int tid  = threadIdx.x;
    const int row0 = blockIdx.x * ROWS;

    // ---- Stage feat = [x | agg/deg | deg | mean_dist | 0-pad] into LDS ----
    for (int i = tid; i < ROWS * KF; i += 256) {
        int r = i / KF, c = i - r * KF;
        int g = row0 + r;
        if (g >= Ntot) g = Ntot - 1;
        float dg  = degrees[g];
        float inv = 1.0f / fmaxf(dg, 1.0f);
        float v;
        if (c < DD)               v = x[g * DD + c];
        else if (c < 2 * DD)      v = agg[g * DD + (c - DD)] * inv;
        else if (c == 2 * DD)     v = dg;
        else if (c == 2 * DD + 1) v = dsum[g] * inv;
        else                      v = 0.0f;
        featbf[i] = (bf16_t)v;
    }
    __syncthreads();

    const int wave = tid >> 5;
    const int lane = tid & 31;
    const int m    = lane & 15;   // A row / C col within tile
    const int half = lane >> 4;   // K-half selector (16-bit operand layout)

    // ---- GEMM1: [32 x 288] @ [288 x 512] -> h ----
    // Per wave: 4 N-tiles x 2 M-tiles; each B fragment feeds 2 WMMAs.
    v8f acc[4][2] = {};
    #pragma unroll
    for (int ks = 0; ks < KS1; ++ks) {
        v16bf a[2];
        #pragma unroll
        for (int mt = 0; mt < 2; ++mt) {
            const bf16_t* ab =
                &featbf[(mt * 16 + m) * KF + ks * 32 + half * 8];
            v8bf alo = *(const v8bf*)ab;         // K = half*8 .. +7
            v8bf ahi = *(const v8bf*)(ab + 16);  // K = 16+half*8 .. +7
            #pragma unroll
            for (int j = 0; j < 8; ++j) {
                a[mt][j] = alo[j];
                a[mt][j + 8] = ahi[j];
            }
        }
        #pragma unroll
        for (int t = 0; t < 4; ++t) {
            int nt  = wave * 4 + t;
            v16bf b = *(const v16bf*)&w1p[((nt * KS1 + ks) * 32 + lane) * 16];
            #pragma unroll
            for (int mt = 0; mt < 2; ++mt) {
                acc[t][mt] = __builtin_amdgcn_wmma_f32_16x16x32_bf16(
                    false, a[mt], false, b, (short)0, acc[t][mt],
                    false, false);
            }
        }
    }
    // bias + relu -> bf16 h tile in LDS (C layout: row = half*8 + vgpr)
    #pragma unroll
    for (int t = 0; t < 4; ++t) {
        int n      = (wave * 4 + t) * 16 + m;
        float bias = b1[n];
        #pragma unroll
        for (int mt = 0; mt < 2; ++mt) {
            #pragma unroll
            for (int r = 0; r < 8; ++r) {
                float v = fmaxf(acc[t][mt][r] + bias, 0.0f);
                hlds[(mt * 16 + half * 8 + r) * HH + n] = (bf16_t)v;
            }
        }
    }
    __syncthreads();

    // ---- GEMM2: [32 x 512] @ [512 x 128] -> out ----
    // Per wave: 1 N-tile x 2 M-tiles; each B fragment feeds 2 WMMAs.
    v8f acc2[2] = {};
    #pragma unroll
    for (int ks = 0; ks < KS2; ++ks) {
        v16bf a[2];
        #pragma unroll
        for (int mt = 0; mt < 2; ++mt) {
            const bf16_t* ab =
                &hlds[(mt * 16 + m) * HH + ks * 32 + half * 8];
            v8bf alo = *(const v8bf*)ab;
            v8bf ahi = *(const v8bf*)(ab + 16);
            #pragma unroll
            for (int j = 0; j < 8; ++j) {
                a[mt][j] = alo[j];
                a[mt][j + 8] = ahi[j];
            }
        }
        v16bf b = *(const v16bf*)&w2p[((wave * KS2 + ks) * 32 + lane) * 16];
        #pragma unroll
        for (int mt = 0; mt < 2; ++mt) {
            acc2[mt] = __builtin_amdgcn_wmma_f32_16x16x32_bf16(
                false, a[mt], false, b, (short)0, acc2[mt], false, false);
        }
    }
    // bias + residual -> f32 y tile in LDS
    {
        int n      = wave * 16 + m;
        float bias = b2[n];
        #pragma unroll
        for (int mt = 0; mt < 2; ++mt) {
            #pragma unroll
            for (int r = 0; r < 8; ++r) {
                int mr = mt * 16 + half * 8 + r;
                int g  = row0 + mr;
                if (g >= Ntot) g = Ntot - 1;
                ylds[mr * DD + n] = acc2[mt][r] + bias + x[g * DD + n];
            }
        }
    }
    __syncthreads();

    // ---- LayerNorm over D=128: 16 lanes per row, 2 rows per thread ----
    {
        int s = tid & 15;  // sub-lane 0..15
        #pragma unroll
        for (int rh = 0; rh < 2; ++rh) {
            int r = (tid >> 4) + rh * 16;  // row 0..31
            float vals[8], sum = 0.0f, sq = 0.0f;
            #pragma unroll
            for (int j = 0; j < 8; ++j) {
                float v = ylds[r * DD + s + 16 * j];
                vals[j] = v;
                sum += v;
                sq  += v * v;
            }
            // xor-reduce within the 16-lane row group (wave32: in-wave)
            #pragma unroll
            for (int off = 8; off >= 1; off >>= 1) {
                sum += __shfl_xor(sum, off, 32);
                sq  += __shfl_xor(sq,  off, 32);
            }
            float mu  = sum * (1.0f / DD);
            float var = sq * (1.0f / DD) - mu * mu;
            float inv = rsqrtf(var + 1e-5f);
            int g = row0 + r;
            if (g < Ntot) {
                #pragma unroll
                for (int j = 0; j < 8; ++j) {
                    int c = s + 16 * j;
                    out[g * DD + c] = (vals[j] - mu) * inv * gamma[c] + beta[c];
                }
            }
        }
    }
}

// ---------------------------------------------------------------------------
extern "C" void kernel_launch(void* const* d_in, const int* in_sizes, int n_in,
                              void* d_out, int out_size, void* d_ws,
                              size_t ws_size, hipStream_t stream) {
    const float*     x       = (const float*)d_in[0];
    const float*     W1      = (const float*)d_in[1];
    const float*     b1      = (const float*)d_in[2];
    const float*     W2      = (const float*)d_in[3];
    const float*     b2      = (const float*)d_in[4];
    const float*     gamma   = (const float*)d_in[5];
    const float*     beta    = (const float*)d_in[6];
    const long long* ei      = (const long long*)d_in[7];
    const float*     ed      = (const float*)d_in[8];
    const float*     degrees = (const float*)d_in[9];

    const int Ntot = in_sizes[0] / DD;  // 50000
    const int E_   = in_sizes[8];       // 640000

    // Workspace layout (256B aligned): agg | dsum | w1pack | w2pack (~26.2MB)
    char*  wsb = (char*)d_ws;
    size_t off = 0;
    float* agg = (float*)(wsb + off);
    off += (size_t)Ntot * DD * sizeof(float); off = (off + 255) & ~(size_t)255;
    float* dsum = (float*)(wsb + off);
    off += (size_t)Ntot * sizeof(float);      off = (off + 255) & ~(size_t)255;
    bf16_t* w1p = (bf16_t*)(wsb + off);
    off += (size_t)W1P_ELEMS * sizeof(bf16_t); off = (off + 255) & ~(size_t)255;
    bf16_t* w2p = (bf16_t*)(wsb + off);

    zero_f32<<<(Ntot * DD + 255) / 256, 256, 0, stream>>>(agg, Ntot * DD);
    zero_f32<<<(Ntot + 255) / 256, 256, 0, stream>>>(dsum, Ntot);

    pack_weights<<<(W1P_ELEMS + W2P_ELEMS + 255) / 256, 256, 0, stream>>>(
        W1, W2, w1p, w2p);

    long long sthreads = (long long)E_ * 32;
    scatter_edges<<<(int)((sthreads + 255) / 256), 256, 0, stream>>>(
        x, ei, ed, agg, dsum, E_);

    fused_gnn<<<(Ntot + ROWS - 1) / ROWS, 256, 0, stream>>>(
        x, b1, b2, gamma, beta, degrees, agg, dsum, w1p, w2p,
        (float*)d_out, Ntot);
}
